// LECSFormer_66228395705137
// MI455X (gfx1250) — matrix operand order repaired and checked by reading, not compiled, not run on hardware
//
#include <hip/hip_runtime.h>

// ---------------------------------------------------------------------------
// LECSFormer block forward for gfx1250 (MI455X, wave32, WMMA).
// All matmuls use v_wmma_f32_16x16x32_f16 (f16 A/B staged in LDS, f32 accum).
// Memory-bound workload: all large-tensor traffic is vectorized (b128).
// ---------------------------------------------------------------------------

#define B_   16
#define H_   56
#define W_   56
#define L_   (H_ * W_)        // 3136
#define C_   192
#define H3C  (3 * C_)         // 576
#define HID  (4 * C_)         // 768
#define M_   (B_ * L_)        // 50176 rows
#define EPS_ 1e-5f

typedef __attribute__((ext_vector_type(16))) _Float16 v16h;
typedef __attribute__((ext_vector_type(8)))  float    v8f;
typedef __attribute__((ext_vector_type(4)))  _Float16 h4;
typedef __attribute__((ext_vector_type(8)))  _Float16 h8;

__device__ __forceinline__ int lane_id() { return (int)(threadIdx.x & 31u); }
__device__ __forceinline__ int wave_id() { return (int)(threadIdx.x >> 5u); }

__device__ __forceinline__ float gelu_f(float v) {
  // exact erf GELU (torch nn.GELU default)
  return 0.5f * v * (1.0f + erff(v * 0.70710678118654752f));
}

__device__ __forceinline__ v8f wmma_f16(v16h a, v16h b, v8f c) {
  // D = A(16x32) * B(32x16) + C, f32 accumulate
  return __builtin_amdgcn_wmma_f32_16x16x32_f16(
      /*neg_a=*/false, a, /*neg_b=*/false, b,
      /*c_mod=*/(short)0, c, /*reuse_a=*/false, /*reuse_b=*/false);
}

// A fragment (16x32 f16) from row-major [16][stride] f16.
// lane l: M = l&15 ; element e: K = (l>>4)*16 + e
__device__ __forceinline__ v16h frag_ld_a(const _Float16* p, int stride) {
  const int l = lane_id();
  const _Float16* r = p + (l & 15) * stride + ((l >> 4) << 4);
  v16h f;
#pragma unroll
  for (int e = 0; e < 16; ++e) f[e] = r[e];
  return f;
}

// B fragment (32x16 K-by-N f16) from row-major [N][stride] storage, i.e.
// source holds B(k,n) at src[n*stride + k].  lane l: N = l&15,
// element e: K = (l>>4)*16 + e   (contiguous k order, matches frag_ld_a)
__device__ __forceinline__ v16h frag_ld_b_nk(const _Float16* p, int stride) {
  const int l = lane_id();
  const _Float16* r = p + (l & 15) * stride + ((l >> 4) << 4);
  v16h f;
#pragma unroll
  for (int e = 0; e < 16; ++e) f[e] = r[e];
  return f;
}

// B fragment (32x16 K-by-N f16) from row-major [K][stride] storage.
__device__ __forceinline__ v16h frag_ld_b_kn(const _Float16* p, int stride) {
  const int l = lane_id();
  const int n  = l & 15;
  const int k0 = (l >> 4) << 4;
  v16h f;
#pragma unroll
  for (int e = 0; e < 16; ++e) f[e] = p[(k0 + e) * stride + n];
  return f;
}

// store 16x16 f32 C tile into row-major [16][stride]
__device__ __forceinline__ void frag_st_c(float* p, int stride, v8f c) {
  const int l  = lane_id();
  const int n  = l & 15;
  const int m0 = (l >> 4) << 3;
#pragma unroll
  for (int i = 0; i < 8; ++i) p[(m0 + i) * stride + n] = c[i];
}

__device__ __forceinline__ h4 cvt4(float4 v) {
  h4 o;
  o[0] = (_Float16)v.x; o[1] = (_Float16)v.y;
  o[2] = (_Float16)v.z; o[3] = (_Float16)v.w;
  return o;
}

// ---------------------------------------------------------------------------
// f32 -> f16 weight conversion (vectorized, n % 4 == 0)
// ---------------------------------------------------------------------------
__global__ void k_f32_to_f16(const float* __restrict__ src,
                             _Float16* __restrict__ dst, int n4) {
  int i = blockIdx.x * blockDim.x + threadIdx.x;
  if (i < n4) {
    float4 v = reinterpret_cast<const float4*>(src)[i];
    reinterpret_cast<h4*>(dst)[i] = cvt4(v);
  }
}

// ---------------------------------------------------------------------------
// depthwise 3x3 SAME conv + eval BN + exact GELU (+optional residual add)
// layout: [B, L, C] channel-fastest; weights (C,1,3,3); 4 channels / thread
// ---------------------------------------------------------------------------
__global__ __launch_bounds__(256)
void k_dwconv_bn_gelu(const float* __restrict__ in, const float* __restrict__ w,
                      const float* __restrict__ cb, const float* __restrict__ g,
                      const float* __restrict__ bb, float* __restrict__ out,
                      int C, int addResidual) {
  const int cw = C >> 2;                       // channel float4 groups
  int idx = blockIdx.x * 256 + threadIdx.x;    // chunk index
  int total = B_ * L_ * cw;
  if (idx >= total) return;
  int c4 = (idx % cw) << 2;
  int l  = (idx / cw) % L_;
  int b  = idx / (cw * L_);
  int hh = l / W_, ww = l % W_;
  const float4* in4 = reinterpret_cast<const float4*>(in);

  float a0 = 0.f, a1 = 0.f, a2 = 0.f, a3 = 0.f;
#pragma unroll
  for (int di = -1; di <= 1; ++di) {
#pragma unroll
    for (int dj = -1; dj <= 1; ++dj) {
      int h2 = hh + di, w2 = ww + dj;
      if (h2 >= 0 && h2 < H_ && w2 >= 0 && w2 < W_) {
        float4 v = in4[((b * L_ + h2 * W_ + w2) * C + c4) >> 2];
        int tap = (di + 1) * 3 + (dj + 1);
        a0 += v.x * w[(c4 + 0) * 9 + tap];
        a1 += v.y * w[(c4 + 1) * 9 + tap];
        a2 += v.z * w[(c4 + 2) * 9 + tap];
        a3 += v.w * w[(c4 + 3) * 9 + tap];
      }
    }
  }
  const float bnf = rsqrtf(1.0f + EPS_);       // BN: mean=0, var=1
  a0 = gelu_f((a0 + cb[c4 + 0]) * (g[c4 + 0] * bnf) + bb[c4 + 0]);
  a1 = gelu_f((a1 + cb[c4 + 1]) * (g[c4 + 1] * bnf) + bb[c4 + 1]);
  a2 = gelu_f((a2 + cb[c4 + 2]) * (g[c4 + 2] * bnf) + bb[c4 + 2]);
  a3 = gelu_f((a3 + cb[c4 + 3]) * (g[c4 + 3] * bnf) + bb[c4 + 3]);
  float4 o;
  if (addResidual) {
    float4 r = in4[idx];
    o.x = r.x + a0; o.y = r.y + a1; o.z = r.z + a2; o.w = r.w + a3;
  } else {
    o.x = a0; o.y = a1; o.z = a2; o.w = a3;
  }
  reinterpret_cast<float4*>(out)[idx] = o;
}

// ---------------------------------------------------------------------------
// GEMM: Out[M,N] = act(A[M,K] * Wt[N,K]^T + bias)   (qkv: no act, l1: GELU)
// 64x64 tile per WG, 8 waves (4 m-tiles x 2x2 n-tiles), K-steps of 32.
// Vectorized staging (b128 loads, b64/b128 LDS stores), coalesced b128 stores.
// ---------------------------------------------------------------------------
__global__ __launch_bounds__(256)
void k_gemm_act(const float* __restrict__ A, const _Float16* __restrict__ Wf,
                const float* __restrict__ bias, float* __restrict__ Out,
                int K, int N, int applyGelu) {
  __shared__ _Float16 As[64 * 40];
  __shared__ _Float16 Bs[64 * 40];
  __shared__ float    Ob[64 * 68];
  const int tid = threadIdx.x;
  const int wid = wave_id();
  const int m0 = blockIdx.x * 64;
  const int n0 = blockIdx.y * 64;
  const int mi = wid & 3;          // m-tile of this wave
  const int nb = (wid >> 2) * 2;   // first n-tile of this wave
  const float4* A4 = reinterpret_cast<const float4*>(A);
  const h8*     W8 = reinterpret_cast<const h8*>(Wf);

  v8f acc0 = {}, acc1 = {};
  for (int k0 = 0; k0 < K; k0 += 32) {
    // A: 64x32 f32 -> f16  (512 float4 chunks)
#pragma unroll
    for (int ch = tid; ch < 512; ch += 256) {
      int r = ch >> 3, k4 = (ch & 7) << 2;
      float4 v = A4[((m0 + r) * K + k0 + k4) >> 2];
      *reinterpret_cast<h4*>(&As[r * 40 + k4]) = cvt4(v);
      if (k0 + 32 < K)
        __builtin_prefetch(&A4[((m0 + r) * K + k0 + 32 + k4) >> 2], 0, 3);
    }
    // B: 64x32 f16 (256 h8 chunks, one per thread)
    {
      int r = tid >> 2, k8 = (tid & 3) << 3;
      *reinterpret_cast<h8*>(&Bs[r * 40 + k8]) = W8[((n0 + r) * K + k0 + k8) >> 3];
    }
    __syncthreads();
    v16h a  = frag_ld_a(As + (mi * 16) * 40, 40);
    v16h b0 = frag_ld_b_nk(Bs + ((nb + 0) * 16) * 40, 40);
    v16h b1 = frag_ld_b_nk(Bs + ((nb + 1) * 16) * 40, 40);
    acc0 = wmma_f16(a, b0, acc0);
    acc1 = wmma_f16(a, b1, acc1);
    __syncthreads();
  }
  // park C tiles in LDS, then coalesced float4 stores with fused bias/GELU
  frag_st_c(Ob + (mi * 16) * 68 + (nb + 0) * 16, 68, acc0);
  frag_st_c(Ob + (mi * 16) * 68 + (nb + 1) * 16, 68, acc1);
  __syncthreads();
#pragma unroll
  for (int ch = tid; ch < 1024; ch += 256) {
    int r = ch >> 4, c4 = (ch & 15) << 2;
    float4 bv = bias ? reinterpret_cast<const float4*>(bias)[(n0 + c4) >> 2]
                     : make_float4(0.f, 0.f, 0.f, 0.f);
    float4 o;
    o.x = Ob[r * 68 + c4 + 0] + bv.x;
    o.y = Ob[r * 68 + c4 + 1] + bv.y;
    o.z = Ob[r * 68 + c4 + 2] + bv.z;
    o.w = Ob[r * 68 + c4 + 3] + bv.w;
    if (applyGelu) {
      o.x = gelu_f(o.x); o.y = gelu_f(o.y);
      o.z = gelu_f(o.z); o.w = gelu_f(o.w);
    }
    reinterpret_cast<float4*>(Out)[((m0 + r) * N + n0 + c4) >> 2] = o;
  }
}

// ---------------------------------------------------------------------------
// GEMM + bias + LayerNorm(192) + residual(s):  Out = res1 + LN(A*W^T+b) + res2
// N fixed = 192 (full row per WG so LN fuses).  64 rows per WG, 8 waves,
// each wave: 1 m-tile x 6 n-tiles.
// ---------------------------------------------------------------------------
__global__ __launch_bounds__(256)
void k_gemm_ln(const float* __restrict__ A, const _Float16* __restrict__ Wf,
               const float* __restrict__ bias,
               const float* __restrict__ lng, const float* __restrict__ lnb,
               const float* __restrict__ res1, const float* __restrict__ res2,
               float* __restrict__ Out, int K) {
  __shared__ _Float16 As[64 * 40];
  __shared__ _Float16 Bs[192 * 40];
  __shared__ float Ob[64 * 193];
  __shared__ float mean_s[64], inv_s[64];
  const int tid = threadIdx.x;
  const int wid = wave_id();
  const int m0  = blockIdx.x * 64;
  const int mi  = wid & 3;
  const int nt0 = (wid >> 2) * 6;
  const float4* A4 = reinterpret_cast<const float4*>(A);
  const h8*     W8 = reinterpret_cast<const h8*>(Wf);

  v8f acc[6];
#pragma unroll
  for (int t = 0; t < 6; ++t) acc[t] = (v8f){};

  for (int k0 = 0; k0 < K; k0 += 32) {
#pragma unroll
    for (int ch = tid; ch < 512; ch += 256) {
      int r = ch >> 3, k4 = (ch & 7) << 2;
      float4 v = A4[((m0 + r) * K + k0 + k4) >> 2];
      *reinterpret_cast<h4*>(&As[r * 40 + k4]) = cvt4(v);
      if (k0 + 32 < K)
        __builtin_prefetch(&A4[((m0 + r) * K + k0 + 32 + k4) >> 2], 0, 3);
    }
#pragma unroll
    for (int ch = tid; ch < 768; ch += 256) {   // 192x32 f16
      int r = ch >> 2, k8 = (ch & 3) << 3;
      *reinterpret_cast<h8*>(&Bs[r * 40 + k8]) = W8[(r * K + k0 + k8) >> 3];
    }
    __syncthreads();
    v16h a = frag_ld_a(As + (mi * 16) * 40, 40);
#pragma unroll
    for (int t = 0; t < 6; ++t) {
      v16h b = frag_ld_b_nk(Bs + ((nt0 + t) * 16) * 40, 40);
      acc[t] = wmma_f16(a, b, acc[t]);
    }
    __syncthreads();
  }
#pragma unroll
  for (int t = 0; t < 6; ++t)
    frag_st_c(Ob + (mi * 16) * 193 + (nt0 + t) * 16, 193, acc[t]);
  __syncthreads();

  // phase 1: per-row mean / inv-std (stride-193 rows -> conflict-free banks)
  if (tid < 64) {
    float s = 0.f, sq = 0.f;
    for (int c = 0; c < 192; ++c) {
      float v = Ob[tid * 193 + c] + bias[c];
      s += v; sq += v * v;
    }
    float mean = s * (1.0f / 192.0f);
    float var  = sq * (1.0f / 192.0f) - mean * mean;
    mean_s[tid] = mean;
    inv_s[tid]  = rsqrtf(var + EPS_);
  }
  __syncthreads();
  // phase 2: vectorized LN apply + residual(s) + b128 stores
#pragma unroll
  for (int ch = tid; ch < 64 * 48; ch += 256) {
    int r = ch / 48, c4 = (ch % 48) << 2;
    int gr = m0 + r;
    float mean = mean_s[r], inv = inv_s[r];
    float4 r1 = reinterpret_cast<const float4*>(res1)[(gr * 192 + c4) >> 2];
    float4 o;
    float v0 = Ob[r * 193 + c4 + 0] + bias[c4 + 0];
    float v1 = Ob[r * 193 + c4 + 1] + bias[c4 + 1];
    float v2 = Ob[r * 193 + c4 + 2] + bias[c4 + 2];
    float v3 = Ob[r * 193 + c4 + 3] + bias[c4 + 3];
    o.x = r1.x + (v0 - mean) * inv * lng[c4 + 0] + lnb[c4 + 0];
    o.y = r1.y + (v1 - mean) * inv * lng[c4 + 1] + lnb[c4 + 1];
    o.z = r1.z + (v2 - mean) * inv * lng[c4 + 2] + lnb[c4 + 2];
    o.w = r1.w + (v3 - mean) * inv * lng[c4 + 3] + lnb[c4 + 3];
    if (res2) {
      float4 r2 = reinterpret_cast<const float4*>(res2)[(gr * 192 + c4) >> 2];
      o.x += r2.x; o.y += r2.y; o.z += r2.z; o.w += r2.w;
    }
    reinterpret_cast<float4*>(Out)[(gr * 192 + c4) >> 2] = o;
  }
}

// ---------------------------------------------------------------------------
// Cross-shaped-window MHSA + LePE for one (half, window, head) per WG.
// half0: windows 56x7 (column strips), half1: 7x56 (row strips); N=392, hd=24.
// ---------------------------------------------------------------------------
#define NTOK 392
#define NPAD 416      // 13 * 32  (K padding for P*V)
#define KSTR 40       // f16 row stride for Kf/Vf/Qf
#define SSTR 416      // row stride for S / P

__global__ __launch_bounds__(256)
void k_attn(const float* __restrict__ qkv,
            const float* __restrict__ lepe0_w, const float* __restrict__ lepe0_b,
            const float* __restrict__ lepe1_w, const float* __restrict__ lepe1_b,
            float* __restrict__ att) {
  __shared__ _Float16 Kf[NPAD * KSTR];
  __shared__ _Float16 Vf[NPAD * KSTR];
  __shared__ _Float16 Qf[16 * KSTR];
  __shared__ float    Sb[16 * SSTR];
  __shared__ _Float16 Pf[16 * SSTR];
  __shared__ float    Op[8 * 256];
  __shared__ float    rowsum[16];
  __shared__ float    red[16][17];

  const int tid  = threadIdx.x;
  const int wid  = wave_id();
  const int id   = blockIdx.x;       // [0, 1024)
  const int hf   = id >> 9;          // half 0/1
  const int win  = (id >> 2) & 127;  // b*8 + strip
  const int head = id & 3;
  const int b    = win >> 3;
  const int strip= win & 7;
  const int Wsp  = hf ? 56 : 7;
  const int Hsp  = hf ? 7 : 56;
  const float qscale = 0.2041241452319315f;  // 24^-0.5
  const float* lw = hf ? lepe1_w : lepe0_w;
  const float* lb = hf ? lepe1_b : lepe0_b;
  const int c96 = head * 24;                 // channel base within half
  const float4* q4 = reinterpret_cast<const float4*>(qkv);

  // ---- stage whole-window K, V into LDS (zero padded), float4 path ----
  for (int i = tid; i < NPAD * KSTR; i += 256) { Kf[i] = (_Float16)0.f; Vf[i] = (_Float16)0.f; }
  __syncthreads();
  for (int i = tid; i < NTOK * 6; i += 256) {
    int t = i / 6, q = (i % 6) << 2;
    int ii = t / Wsp, jj = t % Wsp;
    int h = hf ? (strip * 7 + ii) : ii;
    int w = hf ? jj : (strip * 7 + jj);
    int rowg = (b * L_ + h * W_ + w) * H3C;
    int cbase = hf * 96 + c96;
    float4 kv = q4[(rowg + C_ + cbase + q) >> 2];
    float4 vv = q4[(rowg + 2 * C_ + cbase + q) >> 2];
    *reinterpret_cast<h4*>(&Kf[t * KSTR + q]) = cvt4(kv);
    *reinterpret_cast<h4*>(&Vf[t * KSTR + q]) = cvt4(vv);
  }
  __syncthreads();

  // ---- 25 row blocks of 16 queries ----
  for (int mb = 0; mb < 25; ++mb) {
    const int m0 = mb * 16;
    // load Q block (scaled, zero-padded rows/cols)
    for (int i = tid; i < 16 * KSTR; i += 256) {
      int r = i / KSTR, cc = i % KSTR;
      int t = m0 + r;
      float v = 0.f;
      if (t < NTOK && cc < 24) {
        int ii = t / Wsp, jj = t % Wsp;
        int h = hf ? (strip * 7 + ii) : ii;
        int w = hf ? jj : (strip * 7 + jj);
        int rowg = (b * L_ + h * W_ + w) * H3C;
        v = qkv[rowg + hf * 96 + c96 + cc] * qscale;
      }
      Qf[i] = (_Float16)v;
    }
    __syncthreads();

    // S = Q * K^T  (26 n-tiles of 16 over 8 waves)
    {
      v16h a = frag_ld_a(Qf, KSTR);
      for (int nt = wid; nt < 26; nt += 8) {
        v16h bf = frag_ld_b_nk(Kf + nt * 16 * KSTR, KSTR);
        v8f c = {};
        c = wmma_f16(a, bf, c);
        frag_st_c(Sb + nt * 16, SSTR, c);
      }
    }
    __syncthreads();

    // masked softmax over 392 valid columns; unnormalized P in f16
    {
      int r = tid >> 4, s = tid & 15;
      float lmax = -1e30f;
      for (int c = s; c < NTOK; c += 16) lmax = fmaxf(lmax, Sb[r * SSTR + c]);
      red[r][s] = lmax;
      __syncthreads();
      float rmax = -1e30f;
#pragma unroll
      for (int k = 0; k < 16; ++k) rmax = fmaxf(rmax, red[r][k]);
      float lsum = 0.f;
      for (int c = s; c < NPAD; c += 16) {
        float e = 0.f;
        if (c < NTOK) e = __expf(Sb[r * SSTR + c] - rmax);
        Pf[r * SSTR + c] = (_Float16)e;
        lsum += e;
      }
      __syncthreads();
      red[r][s] = lsum;
      __syncthreads();
      if (s == 0) {
        float tot = 0.f;
#pragma unroll
        for (int k = 0; k < 16; ++k) tot += red[r][k];
        rowsum[r] = tot;
      }
    }
    __syncthreads();

    // O = P * V : 2 n-tiles x 4 K-slots across the 8 waves, partials in LDS
    {
      int ntile = wid & 1;
      int kslot = wid >> 1;
      v8f acc = {};
      for (int kc = kslot; kc < 13; kc += 4) {
        v16h a  = frag_ld_a(Pf + kc * 32, SSTR);
        v16h bf = frag_ld_b_kn(Vf + kc * 32 * KSTR + ntile * 16, KSTR);
        acc = wmma_f16(a, bf, acc);
      }
      frag_st_c(Op + wid * 256, 16, acc);
    }
    __syncthreads();

    // epilogue: reduce partials, normalize, add LePE, float4 stores
    for (int i = tid; i < 16 * 6; i += 256) {
      int r = i / 6, d4 = (i % 6) << 2;
      int t = m0 + r;
      if (t < NTOK) {
        int ii = t / Wsp, jj = t % Wsp;
        float inv = 1.0f / rowsum[r];
        float4 o4;
        float* op = &o4.x;
#pragma unroll
        for (int j = 0; j < 4; ++j) {
          int d = d4 + j;
          int nt = d >> 4, nn = d & 15;
          float o = 0.f;
#pragma unroll
          for (int ks = 0; ks < 4; ++ks) o += Op[(ks * 2 + nt) * 256 + r * 16 + nn];
          o *= inv;
          // LePE: depthwise 3x3 over the window's (Hsp, Wsp) spatial dims
          float le = lb[c96 + d];
#pragma unroll
          for (int di = -1; di <= 1; ++di) {
#pragma unroll
            for (int dj = -1; dj <= 1; ++dj) {
              int i2 = ii + di, j2 = jj + dj;
              if (i2 >= 0 && i2 < Hsp && j2 >= 0 && j2 < Wsp)
                le += (float)Vf[(i2 * Wsp + j2) * KSTR + d] *
                      lw[(c96 + d) * 9 + (di + 1) * 3 + (dj + 1)];
            }
          }
          op[j] = o + le;
        }
        int h = hf ? (strip * 7 + ii) : ii;
        int w = hf ? jj : (strip * 7 + jj);
        reinterpret_cast<float4*>(att)[
            ((b * L_ + h * W_ + w) * C_ + hf * 96 + c96 + d4) >> 2] = o4;
      }
    }
    __syncthreads();
  }
}

// ---------------------------------------------------------------------------
// host side
// ---------------------------------------------------------------------------
extern "C" void kernel_launch(void* const* d_in, const int* in_sizes, int n_in,
                              void* d_out, int out_size, void* d_ws, size_t ws_size,
                              hipStream_t stream) {
  (void)in_sizes; (void)n_in; (void)out_size; (void)ws_size;
  const float* x       = (const float*)d_in[0];
  const float* qkv_w   = (const float*)d_in[1];
  const float* proj_w  = (const float*)d_in[2];
  const float* proj_b  = (const float*)d_in[3];
  const float* ln1_g   = (const float*)d_in[4];
  const float* ln1_b   = (const float*)d_in[5];
  const float* lepe0_w = (const float*)d_in[6];
  const float* lepe0_b = (const float*)d_in[7];
  const float* lepe1_w = (const float*)d_in[8];
  const float* lepe1_b = (const float*)d_in[9];
  const float* lem_w   = (const float*)d_in[10];
  const float* lem_b   = (const float*)d_in[11];
  const float* lem_g   = (const float*)d_in[12];
  const float* lem_be  = (const float*)d_in[13];
  const float* nwc_w   = (const float*)d_in[14];
  const float* nwc_b   = (const float*)d_in[15];
  const float* nwc_g   = (const float*)d_in[16];
  const float* nwc_be  = (const float*)d_in[17];
  const float* l1_w    = (const float*)d_in[18];
  const float* l1_b    = (const float*)d_in[19];
  const float* dw_w    = (const float*)d_in[20];
  const float* dw_b    = (const float*)d_in[21];
  const float* dw_g    = (const float*)d_in[22];
  const float* dw_be   = (const float*)d_in[23];
  const float* l2_w    = (const float*)d_in[24];
  const float* l2_b    = (const float*)d_in[25];
  const float* ln2_g   = (const float*)d_in[26];
  const float* ln2_b   = (const float*)d_in[27];

  float* ws = (float*)d_ws;
  // workspace layout (floats), with aliasing of dead regions:
  const size_t SZ_XC  = (size_t)M_ * C_;    //  9,633,792
  const size_t SZ_QKV = (size_t)M_ * H3C;   // 28,901,376
  const size_t SZ_HID = (size_t)M_ * HID;   // 38,535,168
  float* qkvb = ws;                         // dead after attn
  float* lemb = ws + SZ_QKV;                // dead after proj
  float* attb = ws + SZ_QKV + SZ_XC;        // att; later reused as x2
  float* x1b  = attb + SZ_XC;
  float* hb   = ws;                         // aliases qkv+lem span (38.5M)
  float* h2b  = x1b + SZ_XC;
  float* x2b  = attb;
  _Float16* wf16 = (_Float16*)(h2b + SZ_HID);
  _Float16* qkvw16 = wf16;                  // 576*192  = 110592
  _Float16* projw16 = wf16 + 110592;        // 192*192  =  36864
  _Float16* l1w16   = wf16 + 147456;        // 768*192  = 147456
  _Float16* l2w16   = wf16 + 294912;        // 192*768  = 147456

  // 1) weight conversion (tiny; lives in L2 afterwards)
  k_f32_to_f16<<<(110592 / 4 + 255) / 256, 256, 0, stream>>>(qkv_w, qkvw16, 110592 / 4);
  k_f32_to_f16<<<(36864 / 4 + 255) / 256, 256, 0, stream>>>(proj_w, projw16, 36864 / 4);
  k_f32_to_f16<<<(147456 / 4 + 255) / 256, 256, 0, stream>>>(l1_w, l1w16, 147456 / 4);
  k_f32_to_f16<<<(147456 / 4 + 255) / 256, 256, 0, stream>>>(l2_w, l2w16, 147456 / 4);

  const int nXC4 = B_ * L_ * C_ / 4;
  // 2) LEM branch
  k_dwconv_bn_gelu<<<(nXC4 + 255) / 256, 256, 0, stream>>>(
      x, lem_w, lem_b, lem_g, lem_be, lemb, C_, 0);
  // 3) QKV GEMM  [50176 x 192] * [576 x 192]^T
  k_gemm_act<<<dim3(M_ / 64, H3C / 64), 256, 0, stream>>>(
      x, qkvw16, nullptr, qkvb, C_, H3C, 0);
  // 4) dual cross-shaped-window attention (+LePE)
  k_attn<<<1024, 256, 0, stream>>>(qkvb, lepe0_w, lepe0_b, lepe1_w, lepe1_b, attb);
  // 5) proj GEMM + LN1 + (x + att + lem) residual -> x1
  k_gemm_ln<<<M_ / 64, 256, 0, stream>>>(
      attb, projw16, proj_b, ln1_g, ln1_b, x, lemb, x1b, C_);
  // 6) NWC branch (fused residual) -> x2
  k_dwconv_bn_gelu<<<(nXC4 + 255) / 256, 256, 0, stream>>>(
      x1b, nwc_w, nwc_b, nwc_g, nwc_be, x2b, C_, 1);
  // 7) MLP fc1 + GELU -> h
  k_gemm_act<<<dim3(M_ / 64, HID / 64), 256, 0, stream>>>(
      x2b, l1w16, l1_b, hb, C_, HID, 1);
  // 8) hidden depthwise conv + BN + GELU -> h2
  const int nXH4 = B_ * L_ * HID / 4;
  k_dwconv_bn_gelu<<<(nXH4 + 255) / 256, 256, 0, stream>>>(
      hb, dw_w, dw_b, dw_g, dw_be, h2b, HID, 0);
  // 9) MLP fc2 + LN2 + residual(x2) -> out
  k_gemm_ln<<<M_ / 64, 256, 0, stream>>>(
      h2b, l2w16, l2_b, ln2_g, ln2_b, x2b, nullptr, (float*)d_out, HID);
}
